// NeedlemanWunschAligner_45930380264198
// MI455X (gfx1250) — matrix operand order repaired and checked by reading, not compiled, not run on hardware
//
#include <hip/hip_runtime.h>
#include <hip/hip_bf16.h>

#define L_SEQ   768
#define LP      769
#define DIM     512
#define NBATCH  8
#define NEGV    (-1.0e9f)
#define TILE_PITCH 40   // halves per LDS tile row: 32 data + 8 pad (avoids bank conflicts)

typedef __attribute__((ext_vector_type(16))) _Float16 v16h;
typedef __attribute__((ext_vector_type(8)))  float    v8f;

#if defined(__AMDGCN__) && __has_builtin(__builtin_amdgcn_global_load_async_to_lds_b128) && \
    __has_builtin(__builtin_amdgcn_s_wait_asynccnt)
#define USE_ASYNC_LDS 1
typedef int int4v __attribute__((ext_vector_type(4)));
typedef __attribute__((address_space(1))) int4v* as1_i4p;   // global, 16B payload
typedef __attribute__((address_space(3))) int4v* as3_i4p;   // LDS, 16B payload
#else
#define USE_ASYNC_LDS 0
#endif

// ---------------------------------------------------------------------------
// WMMA fragment loaders (CDNA5 16x16x32 f16 layouts, CDNA5 ISA)
// A (16x32, MxK): lane r(<16): halves[0..7]=K hi*8.., halves[8..15]=K 16+hi*8..
// B (32x16, KxN): lane c(<16) holds col c, halves[0..7]=K hi*8.., halves[8..15]=K 16+hi*8..
// ---------------------------------------------------------------------------
__device__ __forceinline__ v16h frag_load_a(const _Float16* lds, int base_row) {
  const int lane = threadIdx.x & 31;
  const int r    = lane & 15;
  const int hi   = lane >> 4;
  const _Float16* p0 = lds + (base_row + r) * TILE_PITCH + hi * 8;
  union { uint4 q; _Float16 h[8]; } u0, u1;
  u0.q = *(const uint4*)(p0);
  u1.q = *(const uint4*)(p0 + 16);
  v16h f;
#pragma unroll
  for (int t = 0; t < 8; ++t) { f[t] = u0.h[t]; f[8 + t] = u1.h[t]; }
  return f;
}

__device__ __forceinline__ v16h frag_load_b(const _Float16* lds, int base_row) {
  const int lane = threadIdx.x & 31;
  const int c    = lane & 15;
  const int hi   = lane >> 4;
  const _Float16* p = lds + (base_row + c) * TILE_PITCH + hi * 8;
  union { uint4 q; _Float16 h[8]; } u0, u1;
  u0.q = *(const uint4*)(p);
  u1.q = *(const uint4*)(p + 16);
  v16h f;
#pragma unroll
  for (int t = 0; t < 8; ++t) { f[t] = u0.h[t]; f[8 + t] = u1.h[t]; }
  return f;
}

__device__ __forceinline__ float softplus_f(float x) {
  return fmaxf(x, 0.f) + log1pf(__expf(-fabsf(x)));
}
__device__ __forceinline__ float logsigmoid_f(float x) {
  return fminf(x, 0.f) - log1pf(__expf(-fabsf(x)));
}

// ---------------------------------------------------------------------------
// Kernel 1: Out(f16)[M,512] = X(f32)[M,512] @ W(f32)[512,512]^T + bias
// 128x128 block tile, K-chunks of 32, 256 threads (8 waves, 2x4 frags each).
// f32->f16 conversion must pass through VGPRs, so this one stays synchronous.
// ---------------------------------------------------------------------------
__global__ __launch_bounds__(256)
void proj_gemm_kernel(const float* __restrict__ X, const float* __restrict__ W,
                      const float* __restrict__ bias, _Float16* __restrict__ Out) {
  __shared__ __align__(16) _Float16 ldsA[128 * TILE_PITCH];
  __shared__ __align__(16) _Float16 ldsB[128 * TILE_PITCH];
  const int n0  = blockIdx.x * 128;
  const int m0  = blockIdx.y * 128;
  const int tid = threadIdx.x;
  const int wid    = tid >> 5;
  const int wave_m = wid & 3;
  const int wave_n = wid >> 2;

  v8f acc[2][4];
#pragma unroll
  for (int a = 0; a < 2; ++a)
#pragma unroll
    for (int b = 0; b < 4; ++b)
#pragma unroll
      for (int q = 0; q < 8; ++q) acc[a][b][q] = 0.0f;

  for (int k0 = 0; k0 < DIM; k0 += 32) {
    __syncthreads();
#pragma unroll
    for (int it = 0; it < 4; ++it) {           // 1024 float4 groups / 256 thr
      const int idx = it * 256 + tid;
      const int row = idx >> 3;
      const int grp = idx & 7;
      const float4 va = *(const float4*)(X + (size_t)(m0 + row) * DIM + k0 + grp * 4);
      _Float16* d = ldsA + row * TILE_PITCH + grp * 4;
      d[0] = (_Float16)va.x; d[1] = (_Float16)va.y;
      d[2] = (_Float16)va.z; d[3] = (_Float16)va.w;
      const float4 vb = *(const float4*)(W + (size_t)(n0 + row) * DIM + k0 + grp * 4);
      _Float16* e = ldsB + row * TILE_PITCH + grp * 4;
      e[0] = (_Float16)vb.x; e[1] = (_Float16)vb.y;
      e[2] = (_Float16)vb.z; e[3] = (_Float16)vb.w;
    }
    __syncthreads();

    v16h af[2], bf[4];
    af[0] = frag_load_a(ldsA, wave_m * 32);
    af[1] = frag_load_a(ldsA, wave_m * 32 + 16);
#pragma unroll
    for (int fn = 0; fn < 4; ++fn) bf[fn] = frag_load_b(ldsB, wave_n * 64 + fn * 16);
#pragma unroll
    for (int fm = 0; fm < 2; ++fm)
#pragma unroll
      for (int fn = 0; fn < 4; ++fn)
        acc[fm][fn] = __builtin_amdgcn_wmma_f32_16x16x32_f16(
            false, af[fm], false, bf[fn], (short)0, acc[fm][fn], false, false);
  }

  const int lane = tid & 31;
  const int cN   = lane & 15;
  const int rB   = (lane >> 4) * 8;
#pragma unroll
  for (int fm = 0; fm < 2; ++fm)
#pragma unroll
    for (int fn = 0; fn < 4; ++fn) {
      const int col  = n0 + wave_n * 64 + fn * 16 + cN;
      const int rowb = m0 + wave_m * 32 + fm * 16 + rB;
      const float bv = bias[col];
#pragma unroll
      for (int r = 0; r < 8; ++r)
        Out[(size_t)(rowb + r) * DIM + col] = (_Float16)(acc[fm][fn][r] + bv);
    }
}

// ---------------------------------------------------------------------------
// Kernel 2: per batch b, Out[i,j] = epilogue( dot(P[b,i,:], Q[b,j,:]) )
// mode 0: softplus (theta), mode 1: log-sigmoid (A). f16 operands, f32 accum.
// Double-buffered: tiles stream global->LDS via GLOBAL_LOAD_ASYNC_TO_LDS_B128
// (ASYNCcnt) while the previous tile's WMMAs run. 4 async ops/wave/tile; async
// loads complete in order, so s_wait_asynccnt 4 isolates the finished tile.
// ---------------------------------------------------------------------------
__global__ __launch_bounds__(256)
void pair_gemm_kernel(const _Float16* __restrict__ P, const _Float16* __restrict__ Q,
                      float* __restrict__ Out, int mode) {
  __shared__ __align__(16) _Float16 ldsA[2][128 * TILE_PITCH];
  __shared__ __align__(16) _Float16 ldsB[2][128 * TILE_PITCH];
  const int n0  = blockIdx.x * 128;
  const int m0  = blockIdx.y * 128;
  const int b   = blockIdx.z;
  const int tid = threadIdx.x;
  const int wid    = tid >> 5;
  const int wave_m = wid & 3;
  const int wave_n = wid >> 2;

  const _Float16* Pb = P + (size_t)b * L_SEQ * DIM;
  const _Float16* Qb = Q + (size_t)b * L_SEQ * DIM;
  float* Ob = Out + (size_t)b * L_SEQ * L_SEQ;

  v8f acc[2][4];
#pragma unroll
  for (int a = 0; a < 2; ++a)
#pragma unroll
    for (int c = 0; c < 4; ++c)
#pragma unroll
      for (int q = 0; q < 8; ++q) acc[a][c][q] = 0.0f;

  auto issue_tile = [&](int kt, int buf) {
#pragma unroll
    for (int it = 0; it < 2; ++it) {           // 512 uint4 groups / 256 thr
      const int idx = it * 256 + tid;
      const int row = idx >> 2;
      const int grp = idx & 3;
      const _Float16* gA = Pb + (size_t)(m0 + row) * DIM + kt * 32 + grp * 8;
      const _Float16* gB = Qb + (size_t)(n0 + row) * DIM + kt * 32 + grp * 8;
      _Float16* lA = &ldsA[buf][row * TILE_PITCH + grp * 8];
      _Float16* lB = &ldsB[buf][row * TILE_PITCH + grp * 8];
#if USE_ASYNC_LDS
      __builtin_amdgcn_global_load_async_to_lds_b128(
          (as1_i4p)(void*)gA, (as3_i4p)(void*)lA, 0, 0);
      __builtin_amdgcn_global_load_async_to_lds_b128(
          (as1_i4p)(void*)gB, (as3_i4p)(void*)lB, 0, 0);
#else
      *(uint4*)lA = *(const uint4*)gA;
      *(uint4*)lB = *(const uint4*)gB;
#endif
    }
  };

  const int NT = DIM / 32;                     // 16 K-tiles
  issue_tile(0, 0);
  for (int kt = 0; kt < NT; ++kt) {
    const int cur = kt & 1;
    if (kt + 1 < NT) issue_tile(kt + 1, cur ^ 1);
#if USE_ASYNC_LDS
    if (kt + 1 < NT) __builtin_amdgcn_s_wait_asynccnt(4);  // tile kt done, kt+1 in flight
    else             __builtin_amdgcn_s_wait_asynccnt(0);
#endif
    __syncthreads();                           // all waves' tile portions visible

    const _Float16* tA = ldsA[cur];
    const _Float16* tB = ldsB[cur];
    v16h af[2], bf[4];
    af[0] = frag_load_a(tA, wave_m * 32);
    af[1] = frag_load_a(tA, wave_m * 32 + 16);
#pragma unroll
    for (int fn = 0; fn < 4; ++fn) bf[fn] = frag_load_b(tB, wave_n * 64 + fn * 16);
#pragma unroll
    for (int fm = 0; fm < 2; ++fm)
#pragma unroll
      for (int fn = 0; fn < 4; ++fn)
        acc[fm][fn] = __builtin_amdgcn_wmma_f32_16x16x32_f16(
            false, af[fm], false, bf[fn], (short)0, acc[fm][fn], false, false);
    __syncthreads();                           // buffer safe to overwrite next iter
  }

  const int lane = tid & 31;
  const int cN   = lane & 15;
  const int rB   = (lane >> 4) * 8;
#pragma unroll
  for (int fm = 0; fm < 2; ++fm)
#pragma unroll
    for (int fn = 0; fn < 4; ++fn) {
      const int col  = n0 + wave_n * 64 + fn * 16 + cN;
      const int rowb = m0 + wave_m * 32 + fm * 16 + rB;
#pragma unroll
      for (int r = 0; r < 8; ++r) {
        float v = acc[fm][fn][r];
        v = (mode == 0) ? softplus_f(v) : logsigmoid_f(v);
        Ob[(size_t)(rowb + r) * L_SEQ + col] = v;
      }
    }
}

// ---------------------------------------------------------------------------
// Kernel 3: NW forward wavefront. One workgroup per batch element.
// V[i,j] = theta[i-1,j-1] + LSE(V[i-1,j-1], A+V[i,j-1], A+V[i-1,j])
// Diagonals rotate through LDS; full V (LP x LP) stored for backward.
// Exact next-diagonal prefetch hides L2 latency behind the barrier + exp/log.
// ---------------------------------------------------------------------------
__global__ __launch_bounds__(768)
void nw_forward_kernel(const float* __restrict__ theta, const float* __restrict__ A,
                       float* __restrict__ V) {
  __shared__ float diag[3][LP];
  const int b   = blockIdx.x;
  const int tid = threadIdx.x;
  const int T   = blockDim.x;
  const float* th = theta + (size_t)b * L_SEQ * L_SEQ;
  const float* a  = A     + (size_t)b * L_SEQ * L_SEQ;
  float* Vb = V + (size_t)b * LP * LP;

  for (int i = tid; i < LP; i += T) {
    diag[0][i] = (i == 0) ? 0.f : NEGV;   // diag k=0
    diag[1][i] = NEGV;                    // diag k=1
    Vb[i]               = (i == 0) ? 0.f : NEGV;  // row 0
    Vb[(size_t)i * LP]  = (i == 0) ? 0.f : NEGV;  // col 0
  }
  __syncthreads();

  for (int k = 2; k <= 2 * L_SEQ; ++k) {
    float*       dk  = diag[k % 3];
    const float* dk1 = diag[(k - 1) % 3];
    const float* dk2 = diag[(k - 2) % 3];
    const int lo = (k - L_SEQ > 1) ? (k - L_SEQ) : 1;
    const int hi = (k - 1 < L_SEQ) ? (k - 1) : L_SEQ;

    // prefetch exactly the theta/A cells this thread touches at step k+1
    if (k < 2 * L_SEQ) {
      const int lo2 = (k + 1 - L_SEQ > 1) ? (k + 1 - L_SEQ) : 1;
      const int hi2 = (k < L_SEQ) ? k : L_SEQ;
      const int i2  = lo2 + tid;
      if (i2 <= hi2) {
        const size_t o2 = (size_t)(i2 - 1) * L_SEQ + (size_t)(k + 1 - i2 - 1);
        __builtin_prefetch(&th[o2], 0, 0);
        __builtin_prefetch(&a[o2], 0, 0);
      }
    }

    for (int i = lo + tid; i <= hi; i += T) {
      const int j = k - i;
      const float m  = dk2[i - 1];                          // V[i-1,j-1]
      const float av = a[(size_t)(i - 1) * L_SEQ + (j - 1)];
      const float x  = av + dk1[i];                         // A + V[i,j-1]
      const float y  = av + dk1[i - 1];                     // A + V[i-1,j]
      const float M  = fmaxf(m, fmaxf(x, y));
      const float v  = th[(size_t)(i - 1) * L_SEQ + (j - 1)] + M +
                       __logf(__expf(m - M) + __expf(x - M) + __expf(y - M));
      dk[i] = v;
      Vb[(size_t)i * LP + j] = v;
    }
    if (tid == 0 && k <= L_SEQ) { dk[0] = NEGV; dk[k] = NEGV; }  // border cells of this diag
    __syncthreads();
  }
}

// ---------------------------------------------------------------------------
// Kernel 4: NW backward wavefront (exact adjoint).
// E[i,j] = w_m(i+1,j+1)E[i+1,j+1] + w_x(i,j+1)E[i,j+1] + w_y(i+1,j)E[i+1,j]
// with softmax weights recomputed from stored V; aln[i-1,j-1] = E[i,j].
// ---------------------------------------------------------------------------
__global__ __launch_bounds__(768)
void nw_backward_kernel(const float* __restrict__ theta, const float* __restrict__ A,
                        const float* __restrict__ V, float* __restrict__ aln) {
  __shared__ float Ebuf[3][LP];
  const int b   = blockIdx.x;
  const int tid = threadIdx.x;
  const int T   = blockDim.x;
  const float* th = theta + (size_t)b * L_SEQ * L_SEQ;
  const float* a  = A     + (size_t)b * L_SEQ * L_SEQ;
  const float* Vb = V + (size_t)b * LP * LP;
  float* alnb = aln + (size_t)b * L_SEQ * L_SEQ;

  for (int i = tid; i < 3 * LP; i += T) (&Ebuf[0][0])[i] = 0.f;
  __syncthreads();

  for (int k = 2 * L_SEQ; k >= 2; --k) {
    float*       Ek  = Ebuf[k % 3];
    const float* Ek1 = Ebuf[(k + 1) % 3];
    const float* Ek2 = Ebuf[(k + 2) % 3];
    const int lo = (k - L_SEQ > 1) ? (k - L_SEQ) : 1;
    const int hi = (k - 1 < L_SEQ) ? (k - 1) : L_SEQ;

    // prefetch next step's hot cells (V center + theta/A row this thread uses)
    if (k > 2) {
      const int lo2 = (k - 1 - L_SEQ > 1) ? (k - 1 - L_SEQ) : 1;
      const int hi2 = (k - 2 < L_SEQ) ? (k - 2) : L_SEQ;
      const int i2  = lo2 + tid;
      if (i2 <= hi2) {
        const int j2 = k - 1 - i2;
        __builtin_prefetch(&Vb[(size_t)i2 * LP + j2], 0, 0);
        __builtin_prefetch(&th[(size_t)(i2 - 1) * L_SEQ + (j2 - 1)], 0, 0);
        __builtin_prefetch(&a[(size_t)(i2 - 1) * L_SEQ + (j2 - 1)], 0, 0);
      }
    }

    for (int i = lo + tid; i <= hi; i += T) {
      const int j = k - i;
      float e;
      if (i == L_SEQ && j == L_SEQ) {
        e = 1.f;
      } else {
        e = 0.f;
        const float Vij = Vb[(size_t)i * LP + j];
        if (i < L_SEQ && j < L_SEQ) {   // diagonal move into (i+1,j+1)
          const float U = Vb[(size_t)(i + 1) * LP + (j + 1)] - th[(size_t)i * L_SEQ + j];
          e += Ek2[i + 1] * __expf(Vij - U);
        }
        if (j < L_SEQ) {                // horizontal move into (i,j+1)
          const float U = Vb[(size_t)i * LP + (j + 1)] - th[(size_t)(i - 1) * L_SEQ + j];
          e += Ek1[i] * __expf(a[(size_t)(i - 1) * L_SEQ + j] + Vij - U);
        }
        if (i < L_SEQ) {                // vertical move into (i+1,j)
          const float U = Vb[(size_t)(i + 1) * LP + j] - th[(size_t)i * L_SEQ + (j - 1)];
          e += Ek1[i + 1] * __expf(a[(size_t)i * L_SEQ + (j - 1)] + Vij - U);
        }
      }
      Ek[i] = e;
      alnb[(size_t)(i - 1) * L_SEQ + (j - 1)] = e;
    }
    __syncthreads();
  }
}

// ---------------------------------------------------------------------------
extern "C" void kernel_launch(void* const* d_in, const int* in_sizes, int n_in,
                              void* d_out, int out_size, void* d_ws, size_t ws_size,
                              hipStream_t stream) {
  (void)in_sizes; (void)n_in; (void)out_size;
  const float* hx = (const float*)d_in[0];
  const float* hy = (const float*)d_in[1];
  const float* Wm = (const float*)d_in[2];
  const float* bm = (const float*)d_in[3];
  const float* Wg = (const float*)d_in[4];
  const float* bg = (const float*)d_in[5];
  float* out = (float*)d_out;

  const size_t nScore = (size_t)NBATCH * L_SEQ * L_SEQ;
  float* aln_out   = out;
  float* theta_out = out + nScore;
  float* A_out     = out + 2 * nScore;

  const size_t projBytes = (size_t)NBATCH * L_SEQ * DIM * sizeof(_Float16); // 6.29 MB each
  const size_t vBytes    = (size_t)NBATCH * LP * LP * sizeof(float);        // 18.9 MB
  if (ws_size < 4 * projBytes + vBytes) return;  // insufficient scratch

  char* ws = (char*)d_ws;
  _Float16* zx16 = (_Float16*)(ws);
  _Float16* gx16 = (_Float16*)(ws + projBytes);
  _Float16* zy16 = (_Float16*)(ws + 2 * projBytes);
  _Float16* gy16 = (_Float16*)(ws + 3 * projBytes);
  float*    Vws  = (float*)(ws + 4 * projBytes);

  // Stage 1: projections (f32 -> f16 activations), WMMA
  dim3 gProj(DIM / 128, (NBATCH * L_SEQ) / 128);   // (4, 48)
  proj_gemm_kernel<<<gProj, 256, 0, stream>>>(hx, Wm, bm, zx16);
  proj_gemm_kernel<<<gProj, 256, 0, stream>>>(hx, Wg, bg, gx16);
  proj_gemm_kernel<<<gProj, 256, 0, stream>>>(hy, Wm, bm, zy16);
  proj_gemm_kernel<<<gProj, 256, 0, stream>>>(hy, Wg, bg, gy16);

  // Stage 2: batched pairwise scores + epilogues, WMMA + async global->LDS
  dim3 gPair(L_SEQ / 128, L_SEQ / 128, NBATCH);    // (6, 6, 8)
  pair_gemm_kernel<<<gPair, 256, 0, stream>>>(zx16, zy16, theta_out, 0);
  pair_gemm_kernel<<<gPair, 256, 0, stream>>>(gx16, gy16, A_out, 1);

  // Stage 3/4: smooth NW forward + exact adjoint (wavefront, one WG per batch)
  nw_forward_kernel<<<NBATCH, 768, 0, stream>>>(theta_out, A_out, Vws);
  nw_backward_kernel<<<NBATCH, 768, 0, stream>>>(theta_out, A_out, Vws, aln_out);
}
